// BasicBlock_Sparse_23751169147194
// MI455X (gfx1250) — matrix operand order repaired and checked by reading, not compiled, not run on hardware
//
#include <hip/hip_runtime.h>
#include <hip/hip_bf16.h>

typedef unsigned short u16;
typedef unsigned int   u32;
typedef u16   u16x8   __attribute__((ext_vector_type(8)));
typedef __bf16 bf16x16 __attribute__((ext_vector_type(16)));
typedef float f32x8   __attribute__((ext_vector_type(8)));

union FragAB {
    bf16x16 bf;
    u16x8   us[2];
};
static_assert(sizeof(FragAB) == 32, "frag size");

// ---- problem constants (from reference) ----
#define NB      64
#define CIN     64
#define COUT    128
#define FSET    512
#define H0      56
#define W0      56
#define OHH     28
#define OWW     28
#define OPIX    (OHH*OWW)            // 784
#define NPIX    (NB*OPIX)            // 50176
#define K0TOT   (CIN*9)              // 576
#define K1TOT   (COUT*9)             // 1152
#define NELEM_X (NB*CIN*H0*W0)       // 12,845,056
#define NELEM_O (NB*COUT*OPIX)       // 6,422,528
#define NELEM_A (NB*FSET*OPIX)       // 25,690,112
#define KTOP    16
#define CAP     32
#define BN_EPS  1e-5f

__device__ __forceinline__ u16 f2bf(float f) {
    u32 u = __float_as_uint(f);
    u32 r = u + 0x7fffu + ((u >> 16) & 1u);   // round-to-nearest-even
    return (u16)(r >> 16);
}

// ---------------- generic helpers ----------------
__global__ void k_zero(float* p, int n) {
    int i = blockIdx.x * blockDim.x + threadIdx.x;
    if (i < n) p[i] = 0.0f;
}

__global__ void k_f32_to_bf16(const float* __restrict__ src, u16* __restrict__ dst, int n) {
    int i = blockIdx.x * blockDim.x + threadIdx.x;
    if (i < n) dst[i] = f2bf(src[i]);
}

// Weight convert + K-permute to tap-major GEMM order: dst[f][tap*Cin + c] = w[f][c][tap].
// (A and B use the same K permutation, so the GEMM result is unchanged.)
__global__ void k_wperm_bf16(const float* __restrict__ src, u16* __restrict__ dst,
                             int Cin, int Ktot, int total) {
    int i = blockIdx.x * blockDim.x + threadIdx.x;
    if (i >= total) return;
    int f   = i / Ktot;
    int r   = i - f * Ktot;
    int tap = r / Cin;
    int c   = r - tap * Cin;
    dst[i] = f2bf(src[(f * Cin + c) * 9 + tap]);
}

// ---------------- encoder conv as implicit-GEMM WMMA ----------------
// C[M=FSET filters][N=pixels], K = 9*Cin in tap-major order (tap = kh*3+kw).
// Since Cin is a multiple of the 32-wide K-tile, every K-tile sits inside ONE
// (kh,kw) tap: the tap decomposition is uniform SALU work, each thread gathers
// 16 consecutive channels of one pixel (one bounds test, lane-coalesced global
// loads), and LDS staging is pure b128 traffic.  Double-buffered LDS; 4 waves,
// block tile 64x64, wave tile 32x32 (2x2 v_wmma_f32_16x16x32_bf16 per step).
template <int STRIDE, int Cin, int IH, int IW, int Ktot>
__global__ __launch_bounds__(128)
void k_conv_enc_wmma(const u16* __restrict__ Wbf,   // [FSET][Ktot] bf16 (tap-major K)
                     const u16* __restrict__ Xbf,   // [NB][Cin][IH][IW] bf16
                     float* __restrict__ Aout)      // [NB][FSET][OPIX] f32
{
    constexpr int LROW = 40;          // LDS row stride in u16 (80 B, 16B-aligned rows)
    constexpr int TILE = 64 * LROW;
    constexpr int IHW  = IH * IW;
    __shared__ __align__(16) u16 As[2 * TILE];
    __shared__ __align__(16) u16 Bs[2 * TILE];

    const int t     = threadIdx.x;
    const int fBase = blockIdx.x * 64;
    const int pBase = blockIdx.y * 64;

    const int wave    = t >> 5;
    const int lane    = t & 31;
    const int lane_lo = lane & 15;
    const int hi      = lane >> 4;
    const int wm      = (wave >> 1) * 32;   // wave M offset in block tile
    const int wn      = (wave & 1) * 32;    // wave N offset

    // --- B gather bookkeeping: thread owns pixel pl = t&63, K-half khalf = t>>6 ---
    const int pl    = t & 63;
    const int khalf = t >> 6;
    const int p0    = pBase + pl;
    const int nb0   = p0 / OPIX;
    const int pp0   = p0 - nb0 * OPIX;
    const int oh0   = pp0 / OWW, ow0 = pp0 - oh0 * OWW;
    const int ihB   = oh0 * STRIDE - 1;
    const int iwB   = ow0 * STRIDE - 1;
    const int xpix  = nb0 * Cin * IHW;

    // --- A tile: thread t loads 16 contiguous bf16 (two b128) at row mA, col kA0 ---
    const int  mA   = t >> 1;
    const int  kA0  = (t & 1) * 16;
    const u16* Arow = Wbf + (fBase + mA) * Ktot + kA0;

    u16x8 tmpA[2];
    union { u16 e[16]; u16x8 v[2]; } tmpB;

    auto loadTiles = [&](int kb) {
        // weights: vector loads, contiguous along K
        tmpA[0] = *reinterpret_cast<const u16x8*>(Arow + kb);
        tmpA[1] = *reinterpret_cast<const u16x8*>(Arow + kb + 8);
        // im2col: whole K-tile shares one (kh,kw) tap (uniform), 16 channels/thread.
        // ONE bounds predicate guards the whole 16-load burst (single exec toggle).
        const int tap = kb / Cin;
        const int kh  = tap / 3, kw = tap - kh * 3;
        const int c0  = (kb & (Cin - 1)) + khalf * 16;
        const int ih  = ihB + kh, iw = iwB + kw;
        const int a0  = xpix + (c0 * IH + ih) * IW + iw;
        if ((ih >= 0) & (ih < IH) & (iw >= 0) & (iw < IW)) {
            #pragma unroll
            for (int j = 0; j < 16; ++j) tmpB.e[j] = Xbf[a0 + j * IHW];
        } else {
            #pragma unroll
            for (int j = 0; j < 16; ++j) tmpB.e[j] = 0;
        }
    };
    auto storeTiles = [&](u16* Asw, u16* Bsw) {
        *reinterpret_cast<u16x8*>(Asw + mA * LROW + kA0)            = tmpA[0];
        *reinterpret_cast<u16x8*>(Asw + mA * LROW + kA0 + 8)        = tmpA[1];
        *reinterpret_cast<u16x8*>(Bsw + pl * LROW + khalf * 16)     = tmpB.v[0];
        *reinterpret_cast<u16x8*>(Bsw + pl * LROW + khalf * 16 + 8) = tmpB.v[1];
    };

    f32x8 acc[2][2] = {};

    loadTiles(0);
    storeTiles(As, Bs);
    __syncthreads();

    int buf = 0;
    for (int kb = 0; kb < Ktot; kb += 32) {
        const bool more = (kb + 32) < Ktot;
        if (more) loadTiles(kb + 32);      // global loads in flight during WMMAs

        const u16* Asw = As + buf * TILE;
        const u16* Bsw = Bs + buf * TILE;
        FragAB a[2], b[2];
        #pragma unroll
        for (int mt = 0; mt < 2; ++mt) {
            int m = wm + mt * 16 + lane_lo;
            // A 16x32 bf16 layout: e<8 -> K=hi*8+e ; e>=8 -> K=16+hi*8+(e-8)
            a[mt].us[0] = *reinterpret_cast<const u16x8*>(&Asw[m * LROW + hi * 8]);
            a[mt].us[1] = *reinterpret_cast<const u16x8*>(&Asw[m * LROW + 16 + hi * 8]);
        }
        #pragma unroll
        for (int nt = 0; nt < 2; ++nt) {
            int n = wn + nt * 16 + lane_lo;
            // B 32x16 layout: e<8 -> K=hi*8+e ; e>=8 -> K=16+hi*8+(e-8) (Bs stored [pixel][K])
            b[nt].us[0] = *reinterpret_cast<const u16x8*>(&Bsw[n * LROW + hi * 8]);
            b[nt].us[1] = *reinterpret_cast<const u16x8*>(&Bsw[n * LROW + 16 + hi * 8]);
        }
        #pragma unroll
        for (int mt = 0; mt < 2; ++mt)
            #pragma unroll
            for (int nt = 0; nt < 2; ++nt)
                acc[mt][nt] = __builtin_amdgcn_wmma_f32_16x16x32_bf16(
                    false, a[mt].bf, false, b[nt].bf, (short)0, acc[mt][nt],
                    false, false);

        if (more) storeTiles(As + (buf ^ 1) * TILE, Bs + (buf ^ 1) * TILE);
        __syncthreads();
        buf ^= 1;
    }

    // Epilogue: D 16x16 f32 layout: VGPR r -> M = r + hi*8, N = lane_lo
    #pragma unroll
    for (int nt = 0; nt < 2; ++nt) {
        int p  = pBase + wn + nt * 16 + lane_lo;
        int nb = p / OPIX;
        int pp = p - nb * OPIX;
        #pragma unroll
        for (int mt = 0; mt < 2; ++mt) {
            #pragma unroll
            for (int r = 0; r < 8; ++r) {
                int f = fBase + wm + mt * 16 + hi * 8 + r;
                Aout[(nb * FSET + f) * OPIX + pp] = acc[mt][nt][r];
            }
        }
    }
}

// ---------------- per-pixel top-k over the 512 filterset channels ----------------
__global__ void k_topk(const float* __restrict__ A,   // [NB][FSET][OPIX]
                       u32* __restrict__ cnt, u16* __restrict__ idx,
                       float* __restrict__ val)
{
    int p = blockIdx.x * blockDim.x + threadIdx.x;
    if (p >= NPIX) return;
    int nb = p / OPIX;
    int pp = p - nb * OPIX;
    const int base = (nb * FSET) * OPIX + pp;

    float t[KTOP];
    #pragma unroll
    for (int i = 0; i < KTOP; ++i) t[i] = -__builtin_inff();

    for (int ch = 0; ch < FSET; ++ch) {
        float v = A[base + ch * OPIX];
        if (v > t[KTOP - 1]) {
            #pragma unroll
            for (int i = 0; i < KTOP; ++i) {
                if (v > t[i]) { float tmp = t[i]; t[i] = v; v = tmp; }
            }
        }
    }
    float thresh = t[KTOP - 1];

    u32 c = 0;
    for (int ch = 0; ch < FSET; ++ch) {
        float v = A[base + ch * OPIX];
        if (v >= thresh && c < CAP) {
            idx[p * CAP + c] = (u16)ch;
            val[p * CAP + c] = v;
            ++c;
        }
    }
    cnt[p] = c;
}

// ---------------- exact-adjoint reconstruction + aux MSE (sparsity-exploiting) ----------------
template <int STRIDE>
__global__ void k_recon_aux(const u32* __restrict__ cnt, const u16* __restrict__ idx,
                            const float* __restrict__ val,
                            const float* __restrict__ Wenc,   // [FSET][Cin][3][3] f32
                            const float* __restrict__ Xref,   // [NB][Cin][IH][IW]
                            float* __restrict__ auxAcc,
                            int Cin, int IH, int IW, float invTotal)
{
    __shared__ float red[256];
    int gid   = blockIdx.x * blockDim.x + threadIdx.x;
    int total = NB * Cin * IH * IW;
    float d2  = 0.0f;
    if (gid < total) {
        int iw = gid % IW; int tmp = gid / IW;
        int ih = tmp % IH; tmp /= IH;
        int c  = tmp % Cin;
        int nb = tmp / Cin;
        float acc = 0.0f;
        #pragma unroll
        for (int kh = 0; kh < 3; ++kh) {
            int num = ih + 1 - kh;
            if (num < 0) continue;
            if (STRIDE == 2 && (num & 1)) continue;
            int oh = (STRIDE == 2) ? (num >> 1) : num;
            if (oh >= OHH) continue;
            #pragma unroll
            for (int kw = 0; kw < 3; ++kw) {
                int numw = iw + 1 - kw;
                if (numw < 0) continue;
                if (STRIDE == 2 && (numw & 1)) continue;
                int ow = (STRIDE == 2) ? (numw >> 1) : numw;
                if (ow >= OWW) continue;
                int p  = (nb * OHH + oh) * OWW + ow;
                u32 cc = cnt[p]; if (cc > CAP) cc = CAP;
                for (u32 j = 0; j < cc; ++j) {
                    int f = idx[p * CAP + j];
                    acc += val[p * CAP + j] * Wenc[((f * Cin + c) * 3 + kh) * 3 + kw];
                }
            }
        }
        float d = acc - Xref[gid];
        d2 = d * d;
    }
    red[threadIdx.x] = d2;
    __syncthreads();
    for (int s = blockDim.x >> 1; s > 0; s >>= 1) {
        if ((int)threadIdx.x < s) red[threadIdx.x] += red[threadIdx.x + s];
        __syncthreads();
    }
    if (threadIdx.x == 0) atomicAdd(auxAcc, red[0] * invTotal);
}

// ---------------- 1x1 projection on the k-sparse codes ----------------
__global__ __launch_bounds__(128)
void k_proj(const u32* __restrict__ cnt, const u16* __restrict__ idx,
            const float* __restrict__ val,
            const float* __restrict__ Wp,     // [COUT][FSET]
            float* __restrict__ Out)          // [NB][COUT][OPIX]
{
    __shared__ u16   sidx[CAP];
    __shared__ float sval[CAP];
    int p  = blockIdx.x;
    u32 cc = cnt[p]; if (cc > CAP) cc = CAP;
    if (threadIdx.x < cc) {
        sidx[threadIdx.x] = idx[p * CAP + threadIdx.x];
        sval[threadIdx.x] = val[p * CAP + threadIdx.x];
    }
    __syncthreads();
    int co = threadIdx.x;
    float acc = 0.0f;
    for (u32 j = 0; j < cc; ++j) acc += sval[j] * Wp[co * FSET + sidx[j]];
    int nb = p / OPIX;
    int pp = p - nb * OPIX;
    Out[(nb * COUT + co) * OPIX + pp] = acc;
}

// ---------------- training-mode BN: stats then apply ----------------
__global__ void k_bn_stats(const float* __restrict__ X, float* __restrict__ sums,
                           float* __restrict__ sqs)
{
    __shared__ float s1[256], s2[256];
    int c = blockIdx.x;
    const int M = NB * OPIX;
    float a = 0.0f, b = 0.0f;
    for (int i = threadIdx.x; i < M; i += blockDim.x) {
        int nb = i / OPIX;
        int pp = i - nb * OPIX;
        float v = X[(nb * COUT + c) * OPIX + pp];
        a += v; b += v * v;
    }
    s1[threadIdx.x] = a; s2[threadIdx.x] = b;
    __syncthreads();
    for (int s = blockDim.x >> 1; s > 0; s >>= 1) {
        if ((int)threadIdx.x < s) {
            s1[threadIdx.x] += s1[threadIdx.x + s];
            s2[threadIdx.x] += s2[threadIdx.x + s];
        }
        __syncthreads();
    }
    if (threadIdx.x == 0) { sums[c] = s1[0]; sqs[c] = s2[0]; }
}

// BN apply for layer-0 output, fused with bf16 conversion feeding conv1
__global__ void k_bn_apply_bf16(float* __restrict__ X, u16* __restrict__ Xbf,
                                const float* __restrict__ sums, const float* __restrict__ sqs,
                                const float* __restrict__ g, const float* __restrict__ b)
{
    int i = blockIdx.x * blockDim.x + threadIdx.x;
    if (i >= NELEM_O) return;
    int co = (i / OPIX) % COUT;
    const float M = (float)(NB * OPIX);
    float mean = sums[co] / M;
    float var  = sqs[co] / M - mean * mean;
    float inv  = rsqrtf(var + BN_EPS);
    float y = (X[i] - mean) * inv * g[co] + b[co];
    X[i]   = y;
    Xbf[i] = f2bf(y);
}

// ---------------- 1x1 stride-2 shortcut conv (0.8 GF, scalar) ----------------
__global__ void k_shortcut(const float* __restrict__ X, const float* __restrict__ Wsc,
                           float* __restrict__ Out)
{
    int i = blockIdx.x * blockDim.x + threadIdx.x;
    if (i >= NELEM_O) return;
    int pp = i % OPIX;
    int co = (i / OPIX) % COUT;
    int nb = i / (COUT * OPIX);
    int oh = pp / OWW, ow = pp % OWW;
    int ih = oh * 2,  iw = ow * 2;
    float acc = 0.0f;
    for (int ci = 0; ci < CIN; ++ci)
        acc += X[((nb * CIN + ci) * H0 + ih) * W0 + iw] * Wsc[co * CIN + ci];
    Out[i] = acc;
}

// ---------------- final fuse: relu(BN(out1) + BN(shortcut)) ----------------
__global__ void k_fuse(const float* __restrict__ A1, const float* __restrict__ Sc,
                       const float* __restrict__ s1, const float* __restrict__ q1,
                       const float* __restrict__ g1, const float* __restrict__ b1,
                       const float* __restrict__ s2, const float* __restrict__ q2,
                       const float* __restrict__ g2, const float* __restrict__ b2,
                       float* __restrict__ Out)
{
    int i = blockIdx.x * blockDim.x + threadIdx.x;
    if (i >= NELEM_O) return;
    int co = (i / OPIX) % COUT;
    const float M = (float)(NB * OPIX);
    float m1 = s1[co] / M, v1 = q1[co] / M - m1 * m1;
    float m2 = s2[co] / M, v2 = q2[co] / M - m2 * m2;
    float y = (A1[i] - m1) * rsqrtf(v1 + BN_EPS) * g1[co] + b1[co]
            + (Sc[i] - m2) * rsqrtf(v2 + BN_EPS) * g2[co] + b2[co];
    Out[i] = fmaxf(y, 0.0f);
}

__global__ void k_finalize(const float* __restrict__ aux, float* __restrict__ out, int off)
{
    out[off]     = aux[0];
    out[off + 1] = aux[1];
}

// ---------------- host launch ----------------
extern "C" void kernel_launch(void* const* d_in, const int* in_sizes, int n_in,
                              void* d_out, int out_size, void* d_ws, size_t ws_size,
                              hipStream_t stream)
{
    const float* x       = (const float*)d_in[0];
    const float* w_enc0  = (const float*)d_in[1];
    const float* w_proj0 = (const float*)d_in[2];
    const float* g0      = (const float*)d_in[3];
    const float* b0      = (const float*)d_in[4];
    const float* w_enc1  = (const float*)d_in[5];
    const float* w_proj1 = (const float*)d_in[6];
    const float* g1      = (const float*)d_in[7];
    const float* b1      = (const float*)d_in[8];
    const float* w_sc    = (const float*)d_in[9];
    const float* g_sc    = (const float*)d_in[10];
    const float* b_sc    = (const float*)d_in[11];
    float* out = (float*)d_out;
    (void)in_sizes; (void)n_in;

    char* p = (char*)d_ws;
    auto alloc = [&](size_t bytes) -> char* {
        char* r = p;
        p += (bytes + 255) & ~(size_t)255;
        return r;
    };
    float* Abuf   = (float*)alloc(sizeof(float) * NELEM_A);   // a0/a1 (102.8 MB)
    float* out0   = (float*)alloc(sizeof(float) * NELEM_O);   // layer-0 output
    float* out1r  = (float*)alloc(sizeof(float) * NELEM_O);   // pre-BN layer-1 proj
    float* scr    = (float*)alloc(sizeof(float) * NELEM_O);   // pre-BN shortcut
    u16*   xbf    = (u16*)  alloc(sizeof(u16)   * NELEM_X);
    u16*   out0bf = (u16*)  alloc(sizeof(u16)   * NELEM_O);
    u16*   w0bf   = (u16*)  alloc(sizeof(u16)   * FSET * K0TOT);
    u16*   w1bf   = (u16*)  alloc(sizeof(u16)   * FSET * K1TOT);
    u32*   cnt    = (u32*)  alloc(sizeof(u32)   * NPIX);
    u16*   idx    = (u16*)  alloc(sizeof(u16)   * NPIX * CAP);
    float* val    = (float*)alloc(sizeof(float) * NPIX * CAP);
    float* st     = (float*)alloc(sizeof(float) * 128 * 6);   // sums/sqs x3 BNs
    float* aux    = (float*)alloc(sizeof(float) * 2);
    if ((size_t)(p - (char*)d_ws) > ws_size) return;

    k_zero<<<1, 32, 0, stream>>>(aux, 2);

    const int cvt0 = FSET * K0TOT, cvt1 = FSET * K1TOT;
    k_wperm_bf16<<<(cvt0 + 255) / 256, 256, 0, stream>>>(w_enc0, w0bf, CIN,  K0TOT, cvt0);
    k_wperm_bf16<<<(cvt1 + 255) / 256, 256, 0, stream>>>(w_enc1, w1bf, COUT, K1TOT, cvt1);
    k_f32_to_bf16<<<(NELEM_X + 255) / 256, 256, 0, stream>>>(x, xbf, NELEM_X);

    dim3 cgrid(FSET / 64, NPIX / 64);  // (8, 784)

    // ---- layer 0 ----
    k_conv_enc_wmma<2, CIN, H0, W0, K0TOT><<<cgrid, 128, 0, stream>>>(w0bf, xbf, Abuf);
    k_topk<<<NPIX / 256, 256, 0, stream>>>(Abuf, cnt, idx, val);
    k_recon_aux<2><<<(NELEM_X + 255) / 256, 256, 0, stream>>>(
        cnt, idx, val, w_enc0, x, aux + 0, CIN, H0, W0, 1.0f / (float)NELEM_X);
    k_proj<<<NPIX, 128, 0, stream>>>(cnt, idx, val, w_proj0, out0);
    k_bn_stats<<<COUT, 256, 0, stream>>>(out0, st + 0, st + 128);
    k_bn_apply_bf16<<<(NELEM_O + 255) / 256, 256, 0, stream>>>(out0, out0bf,
                                                               st + 0, st + 128, g0, b0);

    // ---- layer 1 ----
    k_conv_enc_wmma<1, COUT, OHH, OWW, K1TOT><<<cgrid, 128, 0, stream>>>(w1bf, out0bf, Abuf);
    k_topk<<<NPIX / 256, 256, 0, stream>>>(Abuf, cnt, idx, val);
    k_recon_aux<1><<<(NELEM_O + 255) / 256, 256, 0, stream>>>(
        cnt, idx, val, w_enc1, out0, aux + 1, COUT, OHH, OWW, 1.0f / (float)NELEM_O);
    k_proj<<<NPIX, 128, 0, stream>>>(cnt, idx, val, w_proj1, out1r);
    k_bn_stats<<<COUT, 256, 0, stream>>>(out1r, st + 256, st + 384);

    // ---- shortcut + fuse ----
    k_shortcut<<<(NELEM_O + 255) / 256, 256, 0, stream>>>(x, w_sc, scr);
    k_bn_stats<<<COUT, 256, 0, stream>>>(scr, st + 512, st + 640);
    k_fuse<<<(NELEM_O + 255) / 256, 256, 0, stream>>>(out1r, scr,
                                                      st + 256, st + 384, g1, b1,
                                                      st + 512, st + 640, g_sc, b_sc, out);
    k_finalize<<<1, 1, 0, stream>>>(aux, out, out_size - 2);
}
